// Contour_58720792871697
// MI455X (gfx1250) — matrix operand order repaired and checked by reading, not compiled, not run on hardware
//
#include <hip/hip_runtime.h>
#include <hip/hip_bf16.h>
#include <math.h>

typedef _Float16 v16h __attribute__((ext_vector_type(16)));
typedef float    v8f  __attribute__((ext_vector_type(8)));

#define PI_F      3.14159265358979323846f
#define TWO_PI_F  6.28318530717958647692f

// workspace layout (in floats)
//  [0,128)        mn/mx per batch: ws[2b]=mn, ws[2b+1]=mx
//  x1f: 64*512    row sums of rounded-normalized left half  (forward contour)
//  x1r: 64*512    row sums of rounded-normalized right half (reversed contour)
//  rArr: 128*512  radius per (b,half)
//  ttArr:128*512  fixed angle per (b,half)
#define OFF_MNMX 0
#define OFF_X1F  128
#define OFF_X1R  (128 + 64*512)
#define OFF_R    (128 + 2*64*512)
#define OFF_TT   (128 + 2*64*512 + 128*512)

// ---------------------------------------------------------------- kernel 1
__global__ void __launch_bounds__(256) k_minmax(const float* __restrict__ in,
                                                float* __restrict__ ws) {
    const int b = blockIdx.x;
    const float* src = in + ((size_t)b * 2 + 1) * (512 * 512); // channel -1
    float mn = 3.4e38f, mx = -3.4e38f;
    for (int i = threadIdx.x; i < 512 * 512; i += 256) {
        float v = src[i];
        mn = fminf(mn, v);
        mx = fmaxf(mx, v);
    }
    #pragma unroll
    for (int off = 16; off > 0; off >>= 1) {
        mn = fminf(mn, __shfl_xor(mn, off, 32));
        mx = fmaxf(mx, __shfl_xor(mx, off, 32));
    }
    __shared__ float smn[8], smx[8];
    const int lane = threadIdx.x & 31, wv = threadIdx.x >> 5;
    if (lane == 0) { smn[wv] = mn; smx[wv] = mx; }
    __syncthreads();
    if (threadIdx.x == 0) {
        float a = smn[0], c = smx[0];
        #pragma unroll
        for (int i = 1; i < 8; ++i) { a = fminf(a, smn[i]); c = fmaxf(c, smx[i]); }
        ws[OFF_MNMX + 2 * b]     = a;
        ws[OFF_MNMX + 2 * b + 1] = c;
    }
}

// ---------------------------------------------------------------- kernel 2
// One wave per image row: x1f = sum_{w<256} round(norm), x1r = sum_{w>=256}.
__global__ void __launch_bounds__(256) k_rowsum(const float* __restrict__ in,
                                                float* __restrict__ ws) {
    const int wv = threadIdx.x >> 5, lane = threadIdx.x & 31;
    const int row = blockIdx.x * 8 + wv;   // 0..32767
    const int b = row >> 9, h = row & 511;
    const float mn = ws[OFF_MNMX + 2 * b];
    const float mx = ws[OFF_MNMX + 2 * b + 1];
    const float inv = 1.0f / (mx - mn);
    const float* src = in + ((size_t)b * 2 + 1) * (512 * 512) + (size_t)h * 512;
    __builtin_prefetch(src + 512, 0, 1);   // global_prefetch_b8 (next row)
    float lo = 0.f, hi = 0.f;
    #pragma unroll
    for (int i = 0; i < 16; ++i) {
        float v = rintf((src[lane + 32 * i] - mn) * inv); // jnp.round = RNE
        if (i < 8) lo += v; else hi += v;
    }
    #pragma unroll
    for (int off = 16; off > 0; off >>= 1) {
        lo += __shfl_xor(lo, off, 32);
        hi += __shfl_xor(hi, off, 32);
    }
    if (lane == 0) {
        ws[OFF_X1F + b * 512 + h] = lo;
        ws[OFF_X1R + b * 512 + h] = hi;
    }
}

// ---------------------------------------------------------------- kernel 3
// Per (b,half): ythtop/bottom, then r / fixed-theta per scanline.
__global__ void __launch_bounds__(512) k_polar(float* __restrict__ ws) {
    const int bh = blockIdx.x;             // b*2 + half
    const int b = bh >> 1, half = bh & 1;
    const int h = threadIdx.x;             // 0..511
    __shared__ float sTop, sBot;
    if (h == 0) { sTop = 0.f; sBot = 0.f; }
    __syncthreads();
    const float* x1p = ws + (half ? OFF_X1R : OFF_X1F) + b * 512;
    const float x1st = x1p[h];
    // clip(relu(x),0,1): sums are exact small integers -> xaux is exactly 0/1
    const float xaux = fminf(fmaxf(fmaxf(x1st, 0.f), 0.f), 1.f);
    if (h < 256) atomicAdd(&sTop, xaux); else atomicAdd(&sBot, xaux);
    __syncthreads();
    const float top = 256.f - sTop;
    const float bot = 256.f + sBot;
    // y1st = min(relu(h - top) + top, bot) == clamp(h, top, bot)
    const float y  = fminf(fmaxf((float)h, top), bot);
    const float xc = -x1st;            // (256 - x1st) - 256
    const float yc = y - 256.f;
    const float r  = sqrtf(xc * xc + yc * yc);
    float tt = atan2f(yc, xc);
    // _fix_radians: -relu(-(2pi - relu(-t))) + relu(t)
    const float ra = fmaxf(tt, 0.f);
    const float rb = fmaxf(-tt, 0.f);
    tt = -fmaxf(-(TWO_PI_F - rb), 0.f) + ra;
    ws[OFF_R  + bh * 512 + h] = r;
    ws[OFF_TT + bh * 512 + h] = tt;
}

// ---------------------------------------------------------------- kernel 4
// Gaussian angular resample via V_WMMA_F32_16X16X32_F16.
// A (16x32 f16) = exp weights for 16 target angles x 32 source-k.
// B (32x16 f16): column0 = r[k], column1 = 1, rest 0.
// D column0 = numerator, column1 = denominator.
__global__ void __launch_bounds__(256) k_interp(const float* __restrict__ ws,
                                                float* __restrict__ out) {
    const int bh = blockIdx.x;             // b*2 + half
    const int b = bh >> 1, half = bh & 1;

    __shared__ float    tt1s[1536];
    __shared__ _Float16 r16s[1536];
    const float* rA  = ws + OFF_R  + bh * 512;
    const float* ttA = ws + OFF_TT + bh * 512;
    for (int i = threadIdx.x; i < 1536; i += 256) {
        const int j = i & 511;
        tt1s[i] = ttA[j] + TWO_PI_F * (float)((i >> 9) - 1); // -2pi, 0, +2pi
        r16s[i] = (_Float16)rA[j];
    }
    __syncthreads();

    const int lane = threadIdx.x & 31;
    const int wv   = threadIdx.x >> 5;     // 8 waves, 4 angle-tiles each
    const int row  = lane & 15;            // A-matrix M within tile
    const int grp  = lane >> 4;            // lane half -> K grouping
    const int nsel = lane & 15;            // B-matrix column

    float ttn[4];
    #pragma unroll
    for (int t = 0; t < 4; ++t) {
        const int m = (wv * 4 + t) * 16 + row;
        ttn[t] = 0.5f * PI_F + (float)m * (PI_F / 512.0f);
    }

    v8f acc[4];
    {
        v8f z = {};
        #pragma unroll
        for (int t = 0; t < 4; ++t) acc[t] = z;
    }

    for (int ks = 0; ks < 1536; ks += 32) {
        // ---- B: 32x16, lanes 0-15 hold K={0..7,16..23}, lanes 16-31 hold K={8..15,24..31}
        v16h bm;
        #pragma unroll
        for (int i = 0; i < 16; ++i) {
            _Float16 v = (_Float16)0.0f;
            if (nsel == 0)      v = r16s[ks + grp * 8 + i + (i & 8)];
            else if (nsel == 1) v = (_Float16)1.0f;
            bm[i] = v;
        }
        // ---- source angles for this lane's A slots
        // lanes 0-15:  K = {0..7, 16..23};  lanes 16-31: K = {8..15, 24..31}
        float ttv[16];
        const int kb = ks + grp * 8;
        #pragma unroll
        for (int i = 0; i < 8; ++i) {
            ttv[i]     = tt1s[kb + i];
            ttv[8 + i] = tt1s[kb + 16 + i];
        }
        #pragma unroll
        for (int t = 0; t < 4; ++t) {
            v16h am;
            #pragma unroll
            for (int i = 0; i < 16; ++i) {
                const float d  = ttv[i] - ttn[t];
                const float d2 = fminf(d * d, 1.0f);
                am[i] = (_Float16)__expf(-100.0f * d2);
            }
            acc[t] = __builtin_amdgcn_wmma_f32_16x16x32_f16(
                false, am, false, bm, (short)0, acc[t], false, false);
        }
    }

    // ---- extract columns N=0 (num) and N=1 (den) from D layout, emit points
    #pragma unroll
    for (int t = 0; t < 4; ++t) {
        const int tb = (wv * 4 + t) * 16;
        #pragma unroll
        for (int j = 0; j < 8; ++j) {
            const float v   = acc[t][j];
            const float nlo = __shfl(v,  0, 32);  // M=j,   N=0
            const float dlo = __shfl(v,  1, 32);  // M=j,   N=1
            const float nhi = __shfl(v, 16, 32);  // M=j+8, N=0
            const float dhi = __shfl(v, 17, 32);  // M=j+8, N=1
            if (lane < 2) {
                const int   m   = tb + j + (lane == 1 ? 8 : 0);
                const float rn  = (lane == 0) ? nlo / (dlo + 1e-20f)
                                              : nhi / (dhi + 1e-20f);
                const float ang = 0.5f * PI_F + (float)m * (PI_F / 512.0f);
                const float x   = rn * cosf(ang) + 256.0f;
                const float y   = rn * sinf(ang) + 256.0f;
                int n; float xo;
                if (half == 0) { n = m;        xo = x; }
                else           { n = 1023 - m; xo = 512.0f - x; } // H - x, reversed
                out[((size_t)b * 1024 + n) * 2 + 0] = xo;
                out[((size_t)b * 1024 + n) * 2 + 1] = y;
            }
        }
    }
}

// ----------------------------------------------------------------
extern "C" void kernel_launch(void* const* d_in, const int* in_sizes, int n_in,
                              void* d_out, int out_size, void* d_ws, size_t ws_size,
                              hipStream_t stream) {
    (void)in_sizes; (void)n_in; (void)out_size; (void)ws_size;
    const float* in  = (const float*)d_in[0];
    float*       out = (float*)d_out;
    float*       ws  = (float*)d_ws;

    k_minmax<<<64,   256, 0, stream>>>(in, ws);
    k_rowsum<<<4096, 256, 0, stream>>>(in, ws);
    k_polar <<<128,  512, 0, stream>>>(ws);
    k_interp<<<128,  256, 0, stream>>>(ws, out);
}
